// CGCNN_9363028705443
// MI455X (gfx1250) — matrix operand order repaired and checked by reading, not compiled, not run on hardware
//
#include <hip/hip_runtime.h>
#include <hip/hip_bf16.h>

// ---------------------------------------------------------------------------
// CGCNN forward for MI455X (gfx1250): bf16 WMMA GEMMs with weights resident
// in the 320KB WGP LDS, fused gather + gate + atomic scatter.
// Fragment loads use contiguous 128-bit LDS reads (ds_load_b128).
// ---------------------------------------------------------------------------

#define ATOMS   50000
#define EDGES   400000
#define FEA     128            // F
#define NBR     64             // D
#define ZK      320            // 2F + D
#define ZP      328            // padded LDS row stride (bf16 elements)
#define NCONV   3
#define TILE_E  128            // edges per workgroup tile
#define NTILES  (EDGES / TILE_E)   // 3125 exactly

typedef __attribute__((ext_vector_type(16))) __bf16 v16bf;
typedef __attribute__((ext_vector_type(8)))  float  v8f;

union Frag16 { v16bf v; uint4 q[2]; };

// LDS layout (dynamic): z tile [128][ZP] bf16, weights [2][128][ZP] bf16, dst idx [128]
#define SMEM_Z_BYTES   (TILE_E * ZP * 2)
#define SMEM_W_BYTES   (2 * FEA * ZP * 2)
#define CONV_SMEM      (SMEM_Z_BYTES + SMEM_W_BYTES + TILE_E * 4)

__device__ __forceinline__ unsigned short f2bf(float f) {
    union { float f; unsigned u; } c; c.f = f;
    unsigned u = c.u;
    unsigned r = (u + 0x7FFFu + ((u >> 16) & 1u)) >> 16;   // round-to-nearest-even
    return (unsigned short)r;
}

// --------------------------- elementwise kernels ---------------------------

__global__ void cg_init_x(const int* __restrict__ an, const float* __restrict__ emb,
                          float* __restrict__ x, unsigned short* __restrict__ xbf) {
    int idx = blockIdx.x * blockDim.x + threadIdx.x;
    int total = ATOMS * FEA;
    if (idx >= total) return;
    int i = idx >> 7;           // / FEA
    int c = idx & 127;          // % FEA
    float v = emb[an[i] * FEA + c];
    x[idx] = v;
    xbf[idx] = f2bf(v);
}

__global__ void cg_edge_feat(const int* __restrict__ ei, const float* __restrict__ pos,
                             unsigned short* __restrict__ ebf) {
    int e = blockIdx.x * blockDim.x + threadIdx.x;
    if (e >= EDGES) return;
    int s = ei[e];
    int d = ei[EDGES + e];
    float dx = pos[s * 3 + 0] - pos[d * 3 + 0];
    float dy = pos[s * 3 + 1] - pos[d * 3 + 1];
    float dz = pos[s * 3 + 2] - pos[d * 3 + 2];
    float dist = sqrtf(dx * dx + dy * dy + dz * dz);
    const float step = 8.0f / 63.0f;
    #pragma unroll 8
    for (int k = 0; k < NBR; ++k) {
        float diff = dist - step * (float)k;
        float v = __expf(-diff * diff * 25.0f);   // 1/GDF_VAR^2 = 25
        ebf[(size_t)e * NBR + k] = f2bf(v);
    }
}

// wT[l][mat][n][k] = W_{f,s}[l][k][n] in bf16 (B-fragments contiguous in K)
__global__ void cg_convert_w(const float* __restrict__ Wf, const float* __restrict__ Ws,
                             unsigned short* __restrict__ wT) {
    int idx = blockIdx.x * blockDim.x + threadIdx.x;
    int total = NCONV * 2 * FEA * ZK;
    if (idx >= total) return;
    int k   = idx % ZK;
    int n   = (idx / ZK) & 127;
    int mat = (idx / (ZK * FEA)) & 1;
    int l   = idx / (ZK * FEA * 2);
    const float* W = mat ? Ws : Wf;
    wT[idx] = f2bf(W[(l * ZK + k) * FEA + n]);
}

__global__ void cg_zero(float* __restrict__ agg, float* __restrict__ stats) {
    size_t total = (size_t)ATOMS * FEA;
    size_t stride = (size_t)gridDim.x * blockDim.x;
    for (size_t i = blockIdx.x * (size_t)blockDim.x + threadIdx.x; i < total + 2 * FEA; i += stride) {
        if (i < total) agg[i] = 0.0f;
        else stats[i - total] = 0.0f;
    }
}

__global__ void cg_stats(const float* __restrict__ agg, float* __restrict__ stats) {
    int c = threadIdx.x & 127;
    int r0 = blockIdx.x * 2 + (threadIdx.x >> 7);
    float s = 0.0f, s2 = 0.0f;
    for (int r = r0; r < ATOMS; r += gridDim.x * 2) {
        float v = agg[(size_t)r * FEA + c];
        s += v; s2 += v * v;
    }
    atomicAdd(&stats[c], s);
    atomicAdd(&stats[FEA + c], s2);
}

__global__ void cg_update(float* __restrict__ x, unsigned short* __restrict__ xbf,
                          const float* __restrict__ agg, const float* __restrict__ stats,
                          const float* __restrict__ gamma, const float* __restrict__ beta) {
    int idx = blockIdx.x * blockDim.x + threadIdx.x;
    int total = ATOMS * FEA;
    if (idx >= total) return;
    int c = idx & 127;
    const float invN = 1.0f / (float)ATOMS;
    float mean = stats[c] * invN;
    float var  = stats[FEA + c] * invN - mean * mean;
    float inv  = rsqrtf(var + 1e-5f);
    float a = (agg[idx] - mean) * inv * gamma[c] + beta[c];
    float xn = x[idx] + a;
    x[idx] = xn;
    xbf[idx] = f2bf(xn);
}

__global__ void cg_copy_out(const float* __restrict__ x, float* __restrict__ out) {
    int idx = blockIdx.x * blockDim.x + threadIdx.x;
    if (idx < ATOMS * FEA) out[idx] = x[idx];
}

// ------------------------------- conv kernel -------------------------------
// One workgroup = 8 wave32. Weights (gate + core matrices, bf16, transposed
// [n][k]) stay in LDS for the kernel lifetime; grid-stride over 128-edge
// tiles. Each wave owns 16 edges (M=16) and sweeps 8 N-tiles x 10 K-chunks,
// accumulating into 128 f32 VGPRs. All fragments load as ds_load_b128.

__global__ __launch_bounds__(256, 1)
void cg_conv(const unsigned short* __restrict__ xbf,
             const unsigned short* __restrict__ ebf,
             const unsigned short* __restrict__ wT,      // [2][128][320] this layer
             const int* __restrict__ ei,
             const float* __restrict__ bfb,              // b_f[l]
             const float* __restrict__ bsb,              // b_s[l]
             float* __restrict__ agg) {
    extern __shared__ char smem[];
    unsigned short* zl   = (unsigned short*)smem;                         // [128][ZP]
    unsigned short* wl   = (unsigned short*)(smem + SMEM_Z_BYTES);        // [2][128][ZP]
    int*            dstl = (int*)(smem + SMEM_Z_BYTES + SMEM_W_BYTES);    // [128]

    const int tid = threadIdx.x;

    // ---- stage weights once: 256 threads, one [320] row each, 16B chunks ----
    {
        int mat = tid >> 7;
        int n   = tid & 127;
        const uint4* src = (const uint4*)(wT + (size_t)(mat * FEA + n) * ZK);
        uint4*       dst = (uint4*)(wl + (size_t)(mat * FEA + n) * ZP);
        #pragma unroll
        for (int i = 0; i < (ZK * 2) / 16; ++i) dst[i] = src[i];   // 40 chunks
    }
    __syncthreads();

    const int lane = tid & 31;
    const int wv   = tid >> 5;          // wave 0..7 -> edge rows [wv*16, wv*16+16)
    const int half = lane >> 4;         // 0: lanes 0-15, 1: lanes 16-31
    const int lm   = lane & 15;

    for (int tile = blockIdx.x; tile < NTILES; tile += gridDim.x) {
        const int base = tile * TILE_E;

        // ---- gather z tile: 2 threads per edge row ----
        {
            int er   = tid >> 1;                 // 0..127
            int part = tid & 1;
            int e    = base + er;
            if (part == 0) {
                int a = ei[EDGES + e];           // dst atom
                dstl[er] = a;
                const uint4* s = (const uint4*)(xbf + (size_t)a * FEA);
                uint4* d = (uint4*)(zl + (size_t)er * ZP);
                #pragma unroll
                for (int i = 0; i < 16; ++i) d[i] = s[i];          // x[dst] -> cols 0..127
            } else {
                int a = ei[e];                   // src atom
                const uint4* s = (const uint4*)(xbf + (size_t)a * FEA);
                uint4* d = (uint4*)(zl + (size_t)er * ZP + FEA);
                #pragma unroll
                for (int i = 0; i < 16; ++i) d[i] = s[i];          // x[src] -> cols 128..255
                const uint4* s2 = (const uint4*)(ebf + (size_t)e * NBR);
                uint4* d2 = (uint4*)(zl + (size_t)er * ZP + 2 * FEA);
                #pragma unroll
                for (int i = 0; i < 8; ++i) d2[i] = s2[i];         // e -> cols 256..319
            }
            // prefetch next tile's edge indices into cache (global_prefetch_b8)
            int nbase = base + (int)gridDim.x * TILE_E;
            if (nbase < EDGES && tid < TILE_E) {
                __builtin_prefetch(&ei[nbase + tid], 0, 0);
                __builtin_prefetch(&ei[EDGES + nbase + tid], 0, 0);
            }
        }
        __syncthreads();

        // ---- WMMA: [16 x 320] x [320 x 128] twice (gate + core) ----
        v8f accf[8], accs[8];
        #pragma unroll
        for (int i = 0; i < 8; ++i) {
            #pragma unroll
            for (int j = 0; j < 8; ++j) { accf[i][j] = 0.0f; accs[i][j] = 0.0f; }
        }

        const int rowOff = (wv * 16 + lm) * ZP;
        #pragma unroll 2
        for (int kc = 0; kc < ZK / 32; ++kc) {
            const int kb = kc * 32;
            // A fragment (16x32 bf16): 32 contiguous bytes per lane.
            //   q[0]: K = kb+half*16  .. +7   ; q[1]: K = kb+half*16+8 .. +7
            Frag16 af;
            {
                const uint4* ap = (const uint4*)(zl + rowOff + kb + (half << 4));
                af.q[0] = ap[0];
                af.q[1] = ap[1];            // +8 bf16 = +1 uint4
            }
            #pragma unroll
            for (int nt = 0; nt < 8; ++nt) {
                const int col = (nt << 4) + lm;
                // B fragment (32x16 bf16): 32 contiguous bytes per lane.
                //   K = kb+half*16 .. +15 at column `col`
                Frag16 bf0, bs0;
                {
                    const uint4* bp = (const uint4*)(wl + (size_t)col * ZP + kb + (half << 4));
                    bf0.q[0] = bp[0];
                    bf0.q[1] = bp[1];
                    const uint4* sp = (const uint4*)(wl + (size_t)(FEA + col) * ZP + kb + (half << 4));
                    bs0.q[0] = sp[0];
                    bs0.q[1] = sp[1];
                }
                accf[nt] = __builtin_amdgcn_wmma_f32_16x16x32_bf16(
                    false, af.v, false, bf0.v, (short)0, accf[nt], false, false);
                accs[nt] = __builtin_amdgcn_wmma_f32_16x16x32_bf16(
                    false, af.v, false, bs0.v, (short)0, accs[nt], false, false);
            }
        }

        // ---- epilogue: gate + scatter-add (C layout: vgpr r -> M=r+8*half, N=lm) ----
        #pragma unroll
        for (int nt = 0; nt < 8; ++nt) {
            const int nn = (nt << 4) + lm;
            const float bfv = bfb[nn];
            const float bsv = bsb[nn];
            #pragma unroll
            for (int r = 0; r < 8; ++r) {
                const int m = r + (half << 3);
                float f = accf[nt][r] + bfv;
                float s = accs[nt][r] + bsv;
                float sig = 1.0f / (1.0f + __expf(-f));
                float sp  = (s > 15.0f) ? s : __logf(1.0f + __expf(s));
                float msg = sig * sp;
                int atom = dstl[wv * 16 + m];
                atomicAdd(&agg[(size_t)atom * FEA + nn], msg);
            }
        }
        __syncthreads();   // z tile reused next iteration
    }
}

// ------------------------------- launcher ----------------------------------

extern "C" void kernel_launch(void* const* d_in, const int* in_sizes, int n_in,
                              void* d_out, int out_size, void* d_ws, size_t ws_size,
                              hipStream_t stream) {
    (void)in_sizes; (void)n_in; (void)out_size; (void)ws_size;

    const int*   an    = (const int*)d_in[0];
    const float* pos   = (const float*)d_in[1];
    const int*   ei    = (const int*)d_in[2];
    /* d_in[3] = batch (unused by reference) */
    const float* emb   = (const float*)d_in[4];
    const float* Wf    = (const float*)d_in[5];
    const float* bfb   = (const float*)d_in[6];
    const float* Ws    = (const float*)d_in[7];
    const float* bsb   = (const float*)d_in[8];
    const float* gamma = (const float*)d_in[9];
    const float* beta  = (const float*)d_in[10];
    float* out = (float*)d_out;

    // workspace carve-up (256B aligned)
    char* ws = (char*)d_ws;
    size_t off = 0;
    auto carve = [&](size_t bytes) { size_t o = off; off += (bytes + 255) & ~(size_t)255; return o; };
    float*          x     = (float*)(ws + carve((size_t)ATOMS * FEA * 4));
    float*          agg   = (float*)(ws + carve((size_t)ATOMS * FEA * 4));
    float*          stats = (float*)(ws + carve(2 * FEA * 4));
    unsigned short* xbf   = (unsigned short*)(ws + carve((size_t)ATOMS * FEA * 2));
    unsigned short* ebf   = (unsigned short*)(ws + carve((size_t)EDGES * NBR * 2));
    unsigned short* wT    = (unsigned short*)(ws + carve((size_t)NCONV * 2 * FEA * ZK * 2));

    // allow > 64KB dynamic LDS for the conv kernel (host-side, capture-safe)
    (void)hipFuncSetAttribute(reinterpret_cast<const void*>(cg_conv),
                              hipFuncAttributeMaxDynamicSharedMemorySize, (int)CONV_SMEM);

    const int total = ATOMS * FEA;
    cg_init_x<<<(total + 255) / 256, 256, 0, stream>>>(an, emb, x, xbf);
    cg_edge_feat<<<(EDGES + 255) / 256, 256, 0, stream>>>(ei, pos, ebf);
    cg_convert_w<<<(NCONV * 2 * FEA * ZK + 255) / 256, 256, 0, stream>>>(Wf, Ws, wT);

    for (int l = 0; l < NCONV; ++l) {
        cg_zero<<<4096, 256, 0, stream>>>(agg, stats);
        cg_conv<<<1024, 256, CONV_SMEM, stream>>>(
            xbf, ebf, wT + (size_t)l * 2 * FEA * ZK, ei,
            bfb + l * FEA, bsb + l * FEA, agg);
        cg_stats<<<256, 256, 0, stream>>>(agg, stats);
        cg_update<<<(total + 255) / 256, 256, 0, stream>>>(
            x, xbf, agg, stats, gamma + l * FEA, beta + l * FEA);
    }
    cg_copy_out<<<(total + 255) / 256, 256, 0, stream>>>(x, out);
}
